// MDNRNN_12060268167933
// MI455X (gfx1250) — compile-verified, matrix operand order checked by reading, not run on hardware
//
#include <hip/hip_runtime.h>
#include <math.h>

// ---------------------------------------------------------------------------
// MDN-RNN (LSTM) for MI455X / gfx1250, wave32, WMMA bf16 path.
//   B=128, T=1000, D_IN=35, H=512 (gates=2048), OUT=480
// Each wave computes a 16x32 output tile: one A fragment, two B fragments,
// two v_wmma_f32_16x16x32_bf16 per k-block (3 loads / WMMA instead of 4).
// ---------------------------------------------------------------------------

typedef __attribute__((ext_vector_type(16))) __bf16 v16bf;
typedef __attribute__((ext_vector_type(8)))  __bf16 v8bf;
typedef __attribute__((ext_vector_type(8)))  float  v8f;

#define Bsz   128
#define Tlen  1000
#define DIN   35
#define Hdim  512
#define G4    2048         // 4*H
#define OUTD  480          // 3*K*Z
#define KB    16           // 512 / 32 k-blocks
#define NT_G  128          // 2048/16 gate n-tiles
#define MT    8            // 128/16  m-tiles
#define NT_O  30           // 480/16  out n-tiles
#define NP_G  64           // gate n-tile pairs (16x32 tiles)
#define NP_O  15           // out  n-tile pairs
#define GATES_BLOCKS (MT * NP_G)   // 512
#define OUT_BLOCKS   (MT * NP_O)   // 120

// workspace byte offsets
#define OFF_WRP 0u                  // Wr packed bf16: 16*128*32*16 = 1,048,576 elem = 2 MB
#define OFF_WDP (2u*1024u*1024u)    // Wd packed bf16: 16*30*32*16  =   245,760 elem
#define OFF_HB  2621440u            // h bf16 row-major 128x512 = 131,072 B
#define OFF_C   2752512u            // c f32 128x512 = 262,144 B
#define OFF_G   3014656u            // gates f32 128x2048 = 1,048,576 B

__device__ __forceinline__ float sigm(float x) { return 1.0f / (1.0f + __expf(-x)); }

// ---------------------------------------------------------------------------
// Pack Wr (512x2048 f32, row-major) into per-lane WMMA B-fragment order.
// Packed elem p: e=p%16, lane=(p/16)%32, nt=(p/512)%128, kb=p/(512*128)
//   K = kb*32 + (lane>>4)*8 + (e<8 ? e : e+8),  N = nt*16 + (lane&15)
// ---------------------------------------------------------------------------
__global__ void mdnrnn_pack_wr(const float* __restrict__ Wr, __bf16* __restrict__ wrp) {
    int p = blockIdx.x * 256 + threadIdx.x;             // 0 .. 1,048,575
    int e    = p & 15;
    int lane = (p >> 4) & 31;
    int nt   = (p >> 9) & 127;
    int kb   = p >> 16;
    int K = kb * 32 + ((lane >> 4) << 3) + (e < 8 ? e : e + 8);
    int N = nt * 16 + (lane & 15);
    wrp[p] = (__bf16)Wr[K * G4 + N];
}

// Same packing for Wd (512x480), nt in [0,30)
__global__ void mdnrnn_pack_wd(const float* __restrict__ Wd, __bf16* __restrict__ wdp) {
    int p = blockIdx.x * 256 + threadIdx.x;             // 0 .. 245,759
    if (p >= KB * NT_O * 32 * 16) return;
    int e    = p & 15;
    int lane = (p >> 4) & 31;
    int nt   = (p >> 9) % NT_O;
    int kb   = p / (512 * NT_O);
    int K = kb * 32 + ((lane >> 4) << 3) + (e < 8 ? e : e + 8);
    int N = nt * 16 + (lane & 15);
    wdp[p] = (__bf16)Wd[K * OUTD + N];
}

__global__ void mdnrnn_init_state(__bf16* __restrict__ hb, float* __restrict__ c) {
    int i = blockIdx.x * 256 + threadIdx.x;             // 0 .. 65,535
    hb[i] = (__bf16)0.0f;
    c[i]  = 0.0f;
}

// ---------------------------------------------------------------------------
// Per-step fused kernel, 632 one-wave blocks:
//   blocks [0,512):  gates(t)  = x_t@Wk + b + h_{t-1}@Wr   (skipped if t>=T)
//   blocks [512,..): out(t-1)  = h_{t-1}@Wd + bd           (skipped if t==0)
// Each wave: 16x32 tile, K=512 swept as 16 k-blocks x 2 WMMA (shared A frag).
// ---------------------------------------------------------------------------
__global__ __launch_bounds__(32)
void mdnrnn_gates_out(const float* __restrict__ inp,   // (B,T,DIN)
                      const float* __restrict__ Wk,    // (DIN, 2048)
                      const float* __restrict__ bvec,  // (2048)
                      const float* __restrict__ bd,    // (480)
                      const __bf16* __restrict__ wrp,  // packed Wr
                      const __bf16* __restrict__ wdp,  // packed Wd
                      const __bf16* __restrict__ hb,   // h_{t-1} bf16 (128x512)
                      float* __restrict__ gates,       // (128x2048)
                      float* __restrict__ out,         // (B*T, 480)
                      int t) {
    __shared__ float sIn[16 * DIN];    // 16 input rows for this m-tile
    __shared__ float sWk[DIN * 32];    // Wk column tile (32 cols)

    const int lane = threadIdx.x;
    const int half = lane >> 4;        // 0 | 1
    const int l15  = lane & 15;
    const int ks0  = half << 3;
    const int blk  = blockIdx.x;

    union AB { v16bf v; v8bf h[2]; };

    if (blk < GATES_BLOCKS) {
        if (t >= Tlen) return;
        const int mt = blk >> 6;       // m-tile 0..7
        const int np = blk & 63;       // n-tile pair 0..63
        const int N0 = np * 32 + l15;
        const int N1 = N0 + 16;

        // Stage x_t rows and Wk column tile through LDS
        for (int i = lane; i < 16 * DIN; i += 32) {
            int r = i / DIN, d = i - r * DIN;
            sIn[i] = inp[((mt * 16 + r) * Tlen + t) * DIN + d];
        }
        for (int i = lane; i < DIN * 32; i += 32) {
            int d = i >> 5, cc = i & 31;
            sWk[i] = Wk[d * G4 + np * 32 + cc];
        }
        __syncthreads();

        // acc init = bias + x_t @ Wk   (K=35 fp32 FMAs from LDS)
        v8f acc0, acc1;
        #pragma unroll
        for (int r = 0; r < 8; ++r) { acc0[r] = bvec[N0]; acc1[r] = bvec[N1]; }
        for (int d = 0; d < DIN; ++d) {
            float w0 = sWk[d * 32 + l15];
            float w1 = sWk[d * 32 + 16 + l15];
            #pragma unroll
            for (int r = 0; r < 8; ++r) {
                float x = sIn[(r + half * 8) * DIN + d];
                acc0[r] += x * w0;
                acc1[r] += x * w1;
            }
        }

        // h_{t-1} @ Wr via bf16 WMMA over K=512; A fragment shared by 2 tiles
        const __bf16* hrow = hb + (mt * 16 + l15) * Hdim;
        const __bf16* bbase = wrp + (size_t)(((np * 2) * 32 + lane) << 4);
        #pragma unroll 4
        for (int kb = 0; kb < KB; ++kb) {
            int ks = kb * 32 + ks0;
            AB a;
            a.h[0] = *(const v8bf*)(hrow + ks);
            a.h[1] = *(const v8bf*)(hrow + ks + 16);
            const __bf16* bp = bbase + (size_t)kb * (NT_G * 32 * 16);
            v16bf bm0 = *(const v16bf*)(bp);
            v16bf bm1 = *(const v16bf*)(bp + 512);
            if (kb + 1 < KB)
                __builtin_prefetch(bbase + (size_t)(kb + 1) * (NT_G * 32 * 16), 0, 3);
            acc0 = __builtin_amdgcn_wmma_f32_16x16x32_bf16(
                false, a.v, false, bm0, (short)0, acc0, false, false);
            acc1 = __builtin_amdgcn_wmma_f32_16x16x32_bf16(
                false, a.v, false, bm1, (short)0, acc1, false, false);
        }

        #pragma unroll
        for (int r = 0; r < 8; ++r) {
            int row = mt * 16 + r + half * 8;
            gates[row * G4 + N0] = acc0[r];
            gates[row * G4 + N1] = acc1[r];
        }
    } else {
        if (t == 0) return;
        const int bo = blk - GATES_BLOCKS;     // 0..119
        const int mt = bo / NP_O;
        const int np = bo % NP_O;
        const int N0 = np * 32 + l15;
        const int N1 = N0 + 16;
        const int tp = t - 1;

        v8f acc0, acc1;
        float b0 = bd[N0], b1 = bd[N1];
        #pragma unroll
        for (int r = 0; r < 8; ++r) { acc0[r] = b0; acc1[r] = b1; }

        const __bf16* hrow = hb + (mt * 16 + l15) * Hdim;
        const __bf16* bbase = wdp + (size_t)(((np * 2) * 32 + lane) << 4);
        #pragma unroll 4
        for (int kb = 0; kb < KB; ++kb) {
            int ks = kb * 32 + ks0;
            AB a;
            a.h[0] = *(const v8bf*)(hrow + ks);
            a.h[1] = *(const v8bf*)(hrow + ks + 16);
            const __bf16* bp = bbase + (size_t)kb * (NT_O * 32 * 16);
            v16bf bm0 = *(const v16bf*)(bp);
            v16bf bm1 = *(const v16bf*)(bp + 512);
            acc0 = __builtin_amdgcn_wmma_f32_16x16x32_bf16(
                false, a.v, false, bm0, (short)0, acc0, false, false);
            acc1 = __builtin_amdgcn_wmma_f32_16x16x32_bf16(
                false, a.v, false, bm1, (short)0, acc1, false, false);
        }

        #pragma unroll
        for (int r = 0; r < 8; ++r) {
            int brow = mt * 16 + r + half * 8;
            out[((size_t)brow * Tlen + tp) * OUTD + N0] = acc0[r];
            out[((size_t)brow * Tlen + tp) * OUTD + N1] = acc1[r];
        }
    }
}

// ---------------------------------------------------------------------------
// LSTM elementwise update: c,h <- gate nonlinearity; writes h as bf16.
// 65536 threads, one per (b, j).
// ---------------------------------------------------------------------------
__global__ __launch_bounds__(256)
void mdnrnn_lstm_update(const float* __restrict__ gates,
                        float* __restrict__ c,
                        __bf16* __restrict__ hb) {
    int idx = blockIdx.x * 256 + threadIdx.x;   // 0..65535
    int b = idx >> 9;
    int j = idx & 511;
    const float* g = gates + (size_t)b * G4;
    float i_ = g[j];
    float f_ = g[j + 512];
    float g_ = g[j + 1024];
    float o_ = g[j + 1536];
    float cp = c[idx];
    float cn = sigm(f_) * cp + sigm(i_) * tanhf(g_);
    float hn = sigm(o_) * tanhf(cn);
    c[idx]  = cn;
    hb[idx] = (__bf16)hn;
}

// ---------------------------------------------------------------------------
extern "C" void kernel_launch(void* const* d_in, const int* in_sizes, int n_in,
                              void* d_out, int out_size, void* d_ws, size_t ws_size,
                              hipStream_t stream) {
    const float* inp  = (const float*)d_in[0];  // (128,1000,35)
    const float* Wk   = (const float*)d_in[1];  // (35,2048)
    const float* Wr   = (const float*)d_in[2];  // (512,2048)
    const float* bvec = (const float*)d_in[3];  // (2048)
    const float* Wd   = (const float*)d_in[4];  // (512,480)
    const float* bd   = (const float*)d_in[5];  // (480)
    float* out = (float*)d_out;                 // (128000, 480)

    char* ws = (char*)d_ws;
    __bf16* wrp   = (__bf16*)(ws + OFF_WRP);
    __bf16* wdp   = (__bf16*)(ws + OFF_WDP);
    __bf16* hb    = (__bf16*)(ws + OFF_HB);
    float*  cbuf  = (float*)(ws + OFF_C);
    float*  gates = (float*)(ws + OFF_G);

    // One-time (per call) packing + state init
    mdnrnn_pack_wr<<<4096, 256, 0, stream>>>(Wr, wrp);       // 1,048,576 elems
    mdnrnn_pack_wd<<<960, 256, 0, stream>>>(Wd, wdp);        // 245,760 elems
    mdnrnn_init_state<<<256, 256, 0, stream>>>(hb, cbuf);    // 65,536 elems

    // Sequential recurrence: step t computes gates(t) and out(t-1) fused,
    // then the elementwise LSTM state update. t==Tlen flushes out(T-1).
    for (int t = 0; t <= Tlen; ++t) {
        mdnrnn_gates_out<<<GATES_BLOCKS + OUT_BLOCKS, 32, 0, stream>>>(
            inp, Wk, bvec, bd, wrp, wdp, hb, gates, out, t);
        if (t < Tlen)
            mdnrnn_lstm_update<<<256, 256, 0, stream>>>(gates, cbuf, hb);
    }
}
